// VanillaLinearAttention_45526653338168
// MI455X (gfx1250) — matrix operand order repaired and checked by reading, not compiled, not run on hardware
//
#include <hip/hip_runtime.h>
#include <hip/hip_bf16.h>

// ---------------------------------------------------------------------------
// Linear attention (B=8, N=4096, C=1024, H=16, D=64) for MI455X / gfx1250.
// Heavy GEMMs on v_wmma_f32_16x16x32_bf16; fragments fetched from LDS as
// paired ds_load_b128; contiguous tile staging via GLOBAL_LOAD_ASYNC_TO_LDS.
// ---------------------------------------------------------------------------

typedef __attribute__((ext_vector_type(16))) __bf16 v16bf;
typedef __attribute__((ext_vector_type(8)))  float  v8f;
typedef __attribute__((ext_vector_type(4)))  __bf16 bf16x4;

#define B_  8
#define N_  4096
#define C_  1024
#define H_  16
#define D_  64
#define SCALE_ 0.125f   // D^-0.5

#define USE_ASYNC 1

union BF8U { uint4 u; __bf16 h[8]; };

// ------------------------------ WMMA helpers -------------------------------

__device__ inline v8f wmma_bf16(v16bf a, v16bf b, v8f c) {
    return __builtin_amdgcn_wmma_f32_16x16x32_bf16(
        false, a, false, b, (short)0, c, false, false);
}

__device__ inline v16bf make_v16(uint4 lo, uint4 hi) {
    struct P { uint4 a, b; } p{lo, hi};
    return __builtin_bit_cast(v16bf, p);
}

// A fragment 16x32 bf16, A[m][k] at p[m*ld + k] (row-major, K contiguous).
__device__ inline v16bf load_a_row(const __bf16* p, int ld) {
    const int lane = threadIdx.x & 31;
    const int hl   = lane >> 4;
    const int m    = lane & 15;
    const __bf16* r = p + m * ld + 8 * hl;
    uint4 lo = *(const uint4*)r;          // k = 8hl .. 8hl+7
    uint4 hi = *(const uint4*)(r + 16);   // k = 16+8hl .. 16+8hl+7
    return make_v16(lo, hi);
}

// B fragment 32x16 bf16 from TRANSPOSED storage: B[k][n] at p[n*ld + k].
__device__ inline v16bf load_b_trans(const __bf16* p, int ld) {
    const int lane = threadIdx.x & 31;
    const int n    = lane & 15;
    const int koff = (lane >> 4) * 8;
    const __bf16* r = p + n * ld + koff;
    uint4 lo = *(const uint4*)r;
    uint4 hi = *(const uint4*)(r + 16);
    return make_v16(lo, hi);
}

// ------------------- async global -> LDS copy (gfx1250) --------------------
// VFLAT async op: per-lane LDS dest offset (VGPR) <- per-lane global address.
// Tracked with ASYNCcnt; publish with s_wait_asynccnt 0 before the barrier.

__device__ inline unsigned lds_off(const void* p) {
    // generic shared pointer = LDS aperture | 32-bit LDS offset (ISA 10.2)
    return (unsigned)(size_t)p;
}

__device__ inline void async_copy_b128(unsigned ldsAddr, const void* gaddr) {
#if USE_ASYNC
    asm volatile("global_load_async_to_lds_b128 %0, %1, off"
                 :: "v"(ldsAddr), "v"(gaddr) : "memory");
#else
    *(uint4*)(size_t)ldsAddr = *(const uint4*)gaddr;   // unused fallback
#endif
}

__device__ inline void wait_async0() {
#if USE_ASYNC
    asm volatile("s_wait_asynccnt 0x0" ::: "memory");
#endif
}

// ------------------------- fp32 -> bf16 conversion -------------------------

__global__ __launch_bounds__(256)
void cvt_f32_bf16(const float* __restrict__ s, __bf16* __restrict__ d, int n4) {
    int i = blockIdx.x * blockDim.x + threadIdx.x;
    if (i < n4) {
        float4 f = ((const float4*)s)[i];
        bf16x4 o;
        o[0] = (__bf16)f.x; o[1] = (__bf16)f.y;
        o[2] = (__bf16)f.z; o[3] = (__bf16)f.w;
        ((bf16x4*)d)[i] = o;
    }
}

// ----------------------- 128x128-tile bf16 WMMA GEMM -----------------------
// 256 threads = 8 waves, 2(M)x4(N); wave patch 64x32 = 8 WMMAs / 32-K step.
// A staged row-major via async-to-LDS; B staged transposed (K contiguous).

#define LDA_ 40
#define LDBT_ 40

__global__ __launch_bounds__(256)
void gemm128_bf16(const __bf16* __restrict__ A, const __bf16* __restrict__ Bw,
                  float* __restrict__ outF, __bf16* __restrict__ outH,
                  const float* __restrict__ bias, int M, int Nn, int K) {
    __shared__ __align__(16) __bf16 lA[2][128 * LDA_];
    __shared__ __align__(16) __bf16 lBt[2][128 * LDBT_];

    const int tid  = threadIdx.x;
    const int wave = tid >> 5;
    const int wm   = wave & 1;
    const int wn   = wave >> 1;
    const int rowBase = blockIdx.y * 128;
    const int colBase = blockIdx.x * 128;
    const int KT = K >> 5;

    // A: 128 rows x 32 cols, two uint4 slots per thread (rows r and r+64).
    const int aR = tid >> 2;
    const int aC = (tid & 3) << 3;
    const __bf16* pa0 = A + (size_t)(rowBase + aR) * K + aC;
    const __bf16* pa1 = pa0 + (size_t)64 * K;
    const int aOfs0 = aR * LDA_ + aC;
    const int aOfs1 = aOfs0 + 64 * LDA_;
    // B: lanes along K rows, 8-col groups along N; two slots (+64 cols).
    const int bR = tid & 31;
    const int bC = (tid >> 5) << 3;
    const __bf16* pb0 = Bw + (size_t)bR * Nn + colBase + bC;
    const __bf16* pb1 = pb0 + 64;
    const size_t bStep = (size_t)32 * Nn;

    auto load_tiles = [&](int buf) {
        async_copy_b128(lds_off(&lA[buf][aOfs0]), pa0);
        async_copy_b128(lds_off(&lA[buf][aOfs1]), pa1);
        BF8U tb0, tb1;
        tb0.u = *(const uint4*)pb0;
        tb1.u = *(const uint4*)pb1;
        __bf16* base = &lBt[buf][bR];
#pragma unroll
        for (int i = 0; i < 8; ++i) {
            base[(bC + i) * LDBT_]      = tb0.h[i];      // lane-contiguous b16
            base[(bC + 64 + i) * LDBT_] = tb1.h[i];
        }
    };

    v8f acc[4][2];
#pragma unroll
    for (int i = 0; i < 4; ++i)
#pragma unroll
        for (int j = 0; j < 2; ++j)
#pragma unroll
            for (int e = 0; e < 8; ++e) acc[i][j][e] = 0.0f;

    load_tiles(0);
    pa0 += 32; pa1 += 32; pb0 += bStep; pb1 += bStep;
    wait_async0();
    __syncthreads();

    for (int kt = 0; kt < KT; ++kt) {
        const int buf = kt & 1;
        if (kt + 1 < KT) {
            load_tiles(buf ^ 1);
            pa0 += 32; pa1 += 32; pb0 += bStep; pb1 += bStep;
            if (kt + 2 < KT) {
                __builtin_prefetch(pa0, 0, 1);
                __builtin_prefetch(pb0, 0, 1);
            }
        }

        v16bf af[4], bfr[2];
#pragma unroll
        for (int i = 0; i < 4; ++i)
            af[i] = load_a_row(&lA[buf][(wm * 64 + 16 * i) * LDA_], LDA_);
#pragma unroll
        for (int j = 0; j < 2; ++j)
            bfr[j] = load_b_trans(&lBt[buf][(wn * 32 + 16 * j) * LDBT_], LDBT_);
#pragma unroll
        for (int i = 0; i < 4; ++i)
#pragma unroll
            for (int j = 0; j < 2; ++j)
                acc[i][j] = wmma_bf16(af[i], bfr[j], acc[i][j]);

        wait_async0();            // publish next buffer's async A tile
        __syncthreads();
    }

    const int lane = tid & 31;
    const int hl   = lane >> 4;
    const int nlow = lane & 15;
#pragma unroll
    for (int i = 0; i < 4; ++i)
#pragma unroll
        for (int j = 0; j < 2; ++j) {
            int gm0 = rowBase + wm * 64 + 16 * i + 8 * hl;
            int gn  = colBase + wn * 32 + 16 * j + nlow;
            float bv = bias ? bias[gn] : 0.0f;
#pragma unroll
            for (int r = 0; r < 8; ++r) {
                float v = acc[i][j][r] + bv;
                if (outF) outF[(size_t)(gm0 + r) * Nn + gn] = v;
                else      outH[(size_t)(gm0 + r) * Nn + gn] = (__bf16)v;
            }
        }
}

// --------------------- q softmax over head_dim (D=64) ----------------------

__global__ __launch_bounds__(256)
void softmax_q(__bf16* __restrict__ q) {
    const int wid  = blockIdx.x * 8 + (threadIdx.x >> 5);
    const int lane = threadIdx.x & 31;
    __bf16* base = q + (size_t)wid * D_;
    float a = (float)base[lane];
    float b = (float)base[lane + 32];
    float m = fmaxf(a, b);
#pragma unroll
    for (int o = 16; o > 0; o >>= 1) m = fmaxf(m, __shfl_xor(m, o, 32));
    float e0 = __expf(a - m), e1 = __expf(b - m);
    float s = e0 + e1;
#pragma unroll
    for (int o = 16; o > 0; o >>= 1) s += __shfl_xor(s, o, 32);
    float w = SCALE_ / s;
    base[lane]      = (__bf16)(e0 * w);
    base[lane + 32] = (__bf16)(e1 * w);
}

// ---------------- k softmax over sequence, two-phase (512 blocks) ----------

#define KSPLIT_ 16
#define KCHUNK_ (N_ / KSPLIT_)   // 256 rows per chunk

__global__ __launch_bounds__(256)
void softmax_k_part(const __bf16* __restrict__ k,
                    float* __restrict__ pm, float* __restrict__ ps) {
    const int b = blockIdx.y, s = blockIdx.z;
    const int c = blockIdx.x * 256 + threadIdx.x;
    const __bf16* base = k + (size_t)b * N_ * C_ + (size_t)s * KCHUNK_ * C_ + c;
    float m = -1e30f;
    for (int n = 0; n < KCHUNK_; ++n)
        m = fmaxf(m, (float)base[(size_t)n * C_]);
    float sum = 0.0f;
    for (int n = 0; n < KCHUNK_; ++n)
        sum += __expf((float)base[(size_t)n * C_] - m);
    size_t idx = ((size_t)b * C_ + c) * KSPLIT_ + s;
    pm[idx] = m;
    ps[idx] = sum;
}

__global__ __launch_bounds__(256)
void softmax_k_norm(__bf16* __restrict__ k,
                    const float* __restrict__ pm, const float* __restrict__ ps) {
    const int b = blockIdx.y, s = blockIdx.z;
    const int c = blockIdx.x * 256 + threadIdx.x;
    size_t idx = ((size_t)b * C_ + c) * KSPLIT_;
    float m = -1e30f;
#pragma unroll
    for (int j = 0; j < KSPLIT_; ++j) m = fmaxf(m, pm[idx + j]);
    float sum = 0.0f;
#pragma unroll
    for (int j = 0; j < KSPLIT_; ++j) sum += ps[idx + j] * __expf(pm[idx + j] - m);
    float inv = 1.0f / sum;
    __bf16* base = k + (size_t)b * N_ * C_ + (size_t)s * KCHUNK_ * C_ + c;
    for (int n = 0; n < KCHUNK_; ++n) {
        float e = __expf((float)base[(size_t)n * C_] - m);
        base[(size_t)n * C_] = (__bf16)(e * inv);
    }
}

// ------------------ kv = K^T V per (b,h): 64x4096x64 WMMA ------------------
// K and V staged TRANSPOSED in LDS ([f][n] / [d][n], n contiguous); output
// written transposed: kvT[b,h][d][f].

#define LDT_ 40

__global__ __launch_bounds__(128)
void kv_kernel(const __bf16* __restrict__ kk, const __bf16* __restrict__ vv,
               __bf16* __restrict__ kvT) {
    const int bh = blockIdx.x;
    const int b  = bh / H_;
    const int h  = bh % H_;
    const int tid  = threadIdx.x;
    const int wave = tid >> 5;

    __shared__ __align__(16) __bf16 kbT[64 * LDT_];
    __shared__ __align__(16) __bf16 vbT[64 * LDT_];

    const int sR = tid & 31;          // n row
    const int sC = (tid >> 5) << 3;   // f/d col base; slot2 adds +32
    const __bf16* gk = kk + ((size_t)(b * N_ + sR)) * C_ + h * D_ + sC;
    const __bf16* gv = vv + ((size_t)(b * N_ + sR)) * C_ + h * D_ + sC;
    const size_t gStep = (size_t)32 * C_;

    v8f acc[4];
#pragma unroll
    for (int t = 0; t < 4; ++t)
#pragma unroll
        for (int e = 0; e < 8; ++e) acc[t][e] = 0.0f;

    for (int n0 = 0; n0 < N_; n0 += 32) {
        BF8U tk0, tk1, tv0, tv1;
        tk0.u = *(const uint4*)gk;
        tk1.u = *(const uint4*)(gk + 32);
        tv0.u = *(const uint4*)gv;
        tv1.u = *(const uint4*)(gv + 32);
        gk += gStep; gv += gStep;
#pragma unroll
        for (int i = 0; i < 8; ++i) {
            kbT[(sC + i) * LDT_ + sR]      = tk0.h[i];
            kbT[(sC + 32 + i) * LDT_ + sR] = tk1.h[i];
            vbT[(sC + i) * LDT_ + sR]      = tv0.h[i];
            vbT[(sC + 32 + i) * LDT_ + sR] = tv1.h[i];
        }
        __syncthreads();
        v16bf af = load_a_row(&kbT[(wave * 16) * LDT_], LDT_);
#pragma unroll
        for (int t = 0; t < 4; ++t) {
            v16bf bf = load_b_trans(&vbT[(t * 16) * LDT_], LDT_);
            acc[t] = wmma_bf16(af, bf, acc[t]);
        }
        __syncthreads();
    }

    const int lane = tid & 31;
    const int hl   = lane >> 4;
    const int nlow = lane & 15;
#pragma unroll
    for (int t = 0; t < 4; ++t)
#pragma unroll
        for (int r = 0; r < 8; ++r) {
            int f = wave * 16 + r + 8 * hl;
            int d = t * 16 + nlow;
            kvT[((size_t)bh * D_ + d) * D_ + f] = (__bf16)acc[t][r];
        }
}

// --------------------- y = q @ kv per (b,h): N x 64 x 64 -------------------

__global__ __launch_bounds__(256)
void y_kernel(const __bf16* __restrict__ q, const __bf16* __restrict__ kvT,
              __bf16* __restrict__ y) {
    const int bh = blockIdx.y;
    const int b  = bh / H_;
    const int h  = bh % H_;
    const int tid  = threadIdx.x;
    const int wave = tid >> 5;
    const int n0   = blockIdx.x * 128 + wave * 16;

    __shared__ __align__(16) __bf16 kvbT[64 * 72];
#pragma unroll
    for (int it = 0; it < 2; ++it) {     // 64x64 = 512 uint4 slots, async
        int slot = it * 256 + tid;
        int r = slot >> 3;
        int c = (slot & 7) << 3;
        async_copy_b128(lds_off(&kvbT[r * 72 + c]),
                        &kvT[((size_t)bh * D_ + r) * D_ + c]);
    }
    wait_async0();
    __syncthreads();

    v8f acc[4];
#pragma unroll
    for (int t = 0; t < 4; ++t)
#pragma unroll
        for (int e = 0; e < 8; ++e) acc[t][e] = 0.0f;

#pragma unroll
    for (int ks = 0; ks < 2; ++ks) {
        v16bf af = load_a_row(&q[((size_t)(b * N_ + n0)) * C_ + h * D_ + ks * 32], C_);
#pragma unroll
        for (int t = 0; t < 4; ++t) {
            v16bf bf = load_b_trans(&kvbT[(t * 16) * 72 + ks * 32], 72);
            acc[t] = wmma_bf16(af, bf, acc[t]);
        }
    }

    const int lane = tid & 31;
    const int hl   = lane >> 4;
    const int nlow = lane & 15;
#pragma unroll
    for (int t = 0; t < 4; ++t)
#pragma unroll
        for (int r = 0; r < 8; ++r) {
            int nrow = n0 + r + 8 * hl;
            int d    = t * 16 + nlow;
            y[((size_t)(b * N_ + nrow)) * C_ + h * D_ + d] = (__bf16)acc[t][r];
        }
}

// ------------------------------ host launcher ------------------------------

extern "C" void kernel_launch(void* const* d_in, const int* in_sizes, int n_in,
                              void* d_out, int out_size, void* d_ws, size_t ws_size,
                              hipStream_t stream) {
    const float* x  = (const float*)d_in[0];
    const float* Wq = (const float*)d_in[1];
    const float* Wk = (const float*)d_in[2];
    const float* Wv = (const float*)d_in[3];
    const float* Wo = (const float*)d_in[4];
    const float* bo = (const float*)d_in[5];
    float* out = (float*)d_out;

    char* ws = (char*)d_ws;
    size_t off = 0;
    const size_t xElems = (size_t)B_ * N_ * C_;
    const size_t wElems = (size_t)C_ * C_;
    __bf16* xb  = (__bf16*)(ws + off); off += xElems * 2;
    __bf16* wqb = (__bf16*)(ws + off); off += wElems * 2;
    __bf16* wkb = (__bf16*)(ws + off); off += wElems * 2;
    __bf16* wvb = (__bf16*)(ws + off); off += wElems * 2;
    __bf16* wob = (__bf16*)(ws + off); off += wElems * 2;
    __bf16* qb  = (__bf16*)(ws + off); off += xElems * 2;
    __bf16* kb2 = (__bf16*)(ws + off); off += xElems * 2;
    __bf16* vb2 = (__bf16*)(ws + off); off += xElems * 2;
    __bf16* kvb = (__bf16*)(ws + off); off += (size_t)B_ * H_ * D_ * D_ * 2;
    __bf16* yb  = (__bf16*)(ws + off); off += xElems * 2;
    float*  pm  = (float*)(ws + off);  off += (size_t)B_ * C_ * KSPLIT_ * 4;
    float*  ps  = (float*)(ws + off);  off += (size_t)B_ * C_ * KSPLIT_ * 4;

    // 1) fp32 -> bf16
    {
        int n4 = (int)(xElems / 4);
        cvt_f32_bf16<<<(n4 + 255) / 256, 256, 0, stream>>>(x, xb, n4);
        int w4 = (int)(wElems / 4);
        cvt_f32_bf16<<<(w4 + 255) / 256, 256, 0, stream>>>(Wq, wqb, w4);
        cvt_f32_bf16<<<(w4 + 255) / 256, 256, 0, stream>>>(Wk, wkb, w4);
        cvt_f32_bf16<<<(w4 + 255) / 256, 256, 0, stream>>>(Wv, wvb, w4);
        cvt_f32_bf16<<<(w4 + 255) / 256, 256, 0, stream>>>(Wo, wob, w4);
    }

    // 2) projections
    dim3 gProj(C_ / 128, (B_ * N_) / 128);
    gemm128_bf16<<<gProj, 256, 0, stream>>>(xb, wqb, nullptr, qb, nullptr,
                                            B_ * N_, C_, C_);
    gemm128_bf16<<<gProj, 256, 0, stream>>>(xb, wkb, nullptr, kb2, nullptr,
                                            B_ * N_, C_, C_);
    gemm128_bf16<<<gProj, 256, 0, stream>>>(xb, wvb, nullptr, vb2, nullptr,
                                            B_ * N_, C_, C_);

    // 3) feature maps
    softmax_q<<<(B_ * N_ * H_) / 8, 256, 0, stream>>>(qb);
    dim3 gK(C_ / 256, B_, KSPLIT_);
    softmax_k_part<<<gK, 256, 0, stream>>>(kb2, pm, ps);
    softmax_k_norm<<<gK, 256, 0, stream>>>(kb2, pm, ps);

    // 4) kv summary (transposed output)
    kv_kernel<<<B_ * H_, 128, 0, stream>>>(kb2, vb2, kvb);

    // 5) apply to q
    dim3 gY(N_ / 128, B_ * H_);
    y_kernel<<<gY, 256, 0, stream>>>(qb, kvb, yb);

    // 6) output projection with bias
    gemm128_bf16<<<gProj, 256, 0, stream>>>(yb, wob, out, nullptr, bo,
                                            B_ * N_, C_, C_);

    (void)in_sizes; (void)n_in; (void)out_size; (void)ws_size;
}
